// Encoder_Cross_62053687492949
// MI455X (gfx1250) — compile-verified, matrix operand order checked by reading, not run on hardware
//
#include <hip/hip_runtime.h>
#include <hip/hip_bf16.h>
#include <math.h>
#include <stdint.h>

// ---------------- problem constants ----------------
#define BQ   8
#define VQ   32
#define PQ   256
#define DQ   512
#define MQ   (BQ*VQ*PQ)     // 65536 tokens
#define MPQ  (BQ*VQ*16)     // 4096 pooled rows
#define ATT_SCALE 0.04419417382415922f   // 1/sqrt(512)

typedef __attribute__((ext_vector_type(16))) __bf16 v16bf;
typedef __attribute__((ext_vector_type(8)))  __bf16 v8bf;
typedef __attribute__((ext_vector_type(8)))  float  v8f;
typedef __attribute__((ext_vector_type(4)))  float  v4f;
typedef __attribute__((ext_vector_type(4)))  int    v4i;

__device__ __forceinline__ __bf16 f2bf(float f) {
  unsigned u = __builtin_bit_cast(unsigned, f);
  unsigned r = u + 0x7FFFu + ((u >> 16) & 1u);
  unsigned short h = (unsigned short)(r >> 16);
  return __builtin_bit_cast(__bf16, h);
}
__device__ __forceinline__ float bf2f(__bf16 b) {
  unsigned short h = __builtin_bit_cast(unsigned short, b);
  return __builtin_bit_cast(float, (unsigned)h << 16);
}
__device__ __forceinline__ float gelu_exact(float v) {
  return 0.5f * v * (1.0f + erff(v * 0.70710678118f));
}

// ---------------- CDNA5 async global->LDS (guarded) ----------------
// Probe result: param0 is 'int __vector(4) __device__ *'  (AS1 int4*),
// so pass AS1/AS3-qualified int4 pointers (global src, lds dst, offset, cpol).
#if __has_builtin(__builtin_amdgcn_global_load_async_to_lds_b128)
#define HAVE_ASYNC_LDS 1
__device__ __forceinline__ void g2l_async16(const void* g, void* l) {
  // low 32 bits of a generic LDS pointer are the LDS byte address (flat aperture rules)
  __builtin_amdgcn_global_load_async_to_lds_b128(
      (__attribute__((address_space(1))) v4i*)(uintptr_t)g,
      (__attribute__((address_space(3))) v4i*)(unsigned)(uintptr_t)l, 0, 0);
}
#else
#define HAVE_ASYNC_LDS 0
__device__ __forceinline__ void g2l_async16(const void*, void*) {}
#endif

#if __has_builtin(__builtin_amdgcn_s_wait_asynccnt)
#define WAIT_ASYNC(n) __builtin_amdgcn_s_wait_asynccnt(n)
#else
#define WAIT_ASYNC(n) asm volatile("s_wait_asynccnt %0" :: "n"(n))
#endif

// load a 16x32 bf16 WMMA fragment from LDS.
// Lane L: row = (L&15), k-base = (L>>4)*8; elements 0..7 = K base..base+7,
// elements 8..15 = K base+16..base+23  (two 16B ds loads).
__device__ __forceinline__ v16bf load_frag(const __bf16* p) {
  v8bf lo = *(const v8bf*)p;
  v8bf hi = *(const v8bf*)(p + 16);
  v16bf f;
#pragma unroll
  for (int i = 0; i < 8; ++i) { f[i] = lo[i]; f[i + 8] = hi[i]; }
  return f;
}

// ---------------- f32 -> bf16 weight conversion ----------------
__global__ __launch_bounds__(256)
void conv_f32_bf16_kernel(const float* __restrict__ in, __bf16* __restrict__ out) {
  int i = (blockIdx.x * 256 + threadIdx.x) * 4;
  v4f f = *(const v4f*)(in + i);
  __bf16 t[4];
#pragma unroll
  for (int j = 0; j < 4; ++j) t[j] = f2bf(f[j]);
  *(unsigned long long*)(out + i) = *(unsigned long long*)t;
}

// ---------------- WMMA GEMM: C[M,N] = epi(A[M,K] @ W[N,K]^T + bias) ----------------
// Block tile 128x128x32, 8 waves, wave tile 32x64 (2x4 wmma). Double-buffered LDS
// fed by CDNA5 async global->LDS loads when available.
// A_F32: A operand is f32 (converted to bf16 while staging into LDS, sync path)
// ACT : 0 none, 1 exact GELU ; RES : 0 none, 1 += f32, 2 += bf16 (stride N)
#define LDA 40  // halves; 80B row stride keeps 16B alignment, skews banks

__device__ __forceinline__ void mma_step(const __bf16* Abase, const __bf16* Bbase,
                                         int wm, int wn, int fr, int fk,
                                         v8f acc[2][4]) {
  v16bf af[2], bfr[4];
#pragma unroll
  for (int i = 0; i < 2; ++i) af[i] = load_frag(Abase + (size_t)(wm + 16 * i + fr) * LDA + fk);
#pragma unroll
  for (int j = 0; j < 4; ++j) bfr[j] = load_frag(Bbase + (size_t)(wn + 16 * j + fr) * LDA + fk);
#pragma unroll
  for (int i = 0; i < 2; ++i)
#pragma unroll
    for (int j = 0; j < 4; ++j)
      acc[i][j] = __builtin_amdgcn_wmma_f32_16x16x32_bf16(
          false, af[i], false, bfr[j], (short)0, acc[i][j], false, false);
}

template<int A_F32, int ACT, int RES>
__global__ __launch_bounds__(256)
void gemm_bf16_kernel(const void* __restrict__ Ap,
                      const __bf16* __restrict__ W,
                      const float* __restrict__ bias,
                      const void* __restrict__ Rp,
                      __bf16* __restrict__ C,
                      int M, int N, int K) {
  __shared__ __bf16 As[2][128][LDA];
  __shared__ __bf16 Bs[2][128][LDA];

  const int tid  = threadIdx.x;
  const int lane = tid & 31;
  const int wave = tid >> 5;
  const int m0 = blockIdx.x * 128;
  const int n0 = blockIdx.y * 128;
  const int wm = (wave & 3) * 32;   // 0,32,64,96
  const int wn = (wave >> 2) * 64;  // 0,64

  v8f acc[2][4] = {};

  const int arow = tid >> 1;        // 0..127 (A and B tiles both 128 rows)
  const int acol = (tid & 1) * 16;  // 0 / 16

  const int fr = lane & 15;
  const int fk = (lane >> 4) * 8;

  if (!A_F32 && HAVE_ASYNC_LDS) {
    // -------- async double-buffered pipeline --------
    const __bf16* Abf = (const __bf16*)Ap;
    const int NK = K >> 5;
    {
      const __bf16* A  = Abf + (size_t)(m0 + arow) * K + acol;
      const __bf16* Wt = W   + (size_t)(n0 + arow) * K + acol;
      g2l_async16(A,      &As[0][arow][acol]);
      g2l_async16(A + 8,  &As[0][arow][acol + 8]);
      g2l_async16(Wt,     &Bs[0][arow][acol]);
      g2l_async16(Wt + 8, &Bs[0][arow][acol + 8]);
    }
    for (int it = 0; it < NK; ++it) {
      const int cur = it & 1;
      if (it + 1 < NK) {
        const int k0 = (it + 1) * 32;
        const __bf16* A  = Abf + (size_t)(m0 + arow) * K + k0 + acol;
        const __bf16* Wt = W   + (size_t)(n0 + arow) * K + k0 + acol;
        g2l_async16(A,      &As[cur ^ 1][arow][acol]);
        g2l_async16(A + 8,  &As[cur ^ 1][arow][acol + 8]);
        g2l_async16(Wt,     &Bs[cur ^ 1][arow][acol]);
        g2l_async16(Wt + 8, &Bs[cur ^ 1][arow][acol + 8]);
        WAIT_ASYNC(4);   // tile `it` complete; next tile still in flight
      } else {
        WAIT_ASYNC(0);
      }
      __syncthreads();
      mma_step(&As[cur][0][0], &Bs[cur][0][0], wm, wn, fr, fk, acc);
      __syncthreads();   // protect buffer `cur` from overwrite at it+2
    }
  } else {
    // -------- sync single-buffered path (also used for f32 A conversion) --------
    for (int k0 = 0; k0 < K; k0 += 32) {
      if (A_F32) {
        const float* A = (const float*)Ap + (size_t)(m0 + arow) * K + k0 + acol;
        v4f f0 = *(const v4f*)(A + 0);
        v4f f1 = *(const v4f*)(A + 4);
        v4f f2 = *(const v4f*)(A + 8);
        v4f f3 = *(const v4f*)(A + 12);
        __bf16 t[16];
#pragma unroll
        for (int i = 0; i < 4; ++i) {
          t[i] = f2bf(f0[i]); t[4 + i] = f2bf(f1[i]);
          t[8 + i] = f2bf(f2[i]); t[12 + i] = f2bf(f3[i]);
        }
        *(v8bf*)&As[0][arow][acol]     = *(v8bf*)&t[0];
        *(v8bf*)&As[0][arow][acol + 8] = *(v8bf*)&t[8];
      } else {
        const __bf16* A = (const __bf16*)Ap + (size_t)(m0 + arow) * K + k0 + acol;
        *(v8bf*)&As[0][arow][acol]     = *(const v8bf*)A;
        *(v8bf*)&As[0][arow][acol + 8] = *(const v8bf*)(A + 8);
      }
      {
        const __bf16* Wt = W + (size_t)(n0 + arow) * K + k0 + acol;
        *(v8bf*)&Bs[0][arow][acol]     = *(const v8bf*)Wt;
        *(v8bf*)&Bs[0][arow][acol + 8] = *(const v8bf*)(Wt + 8);
      }
      __syncthreads();
      mma_step(&As[0][0][0], &Bs[0][0][0], wm, wn, fr, fk, acc);
      __syncthreads();
    }
  }

  // ---- epilogue: bias (+act) (+residual), write bf16 ----
  const int mofs = (lane >> 4) * 8;
#pragma unroll
  for (int tm = 0; tm < 2; ++tm) {
#pragma unroll
    for (int tn = 0; tn < 4; ++tn) {
      const int gn = n0 + wn + tn * 16 + (lane & 15);
      const float bv = bias[gn];
#pragma unroll
      for (int i = 0; i < 8; ++i) {
        const int gm = m0 + wm + tm * 16 + mofs + i;
        float v = acc[tm][tn][i] + bv;
        if (ACT == 1) v = gelu_exact(v);
        if (RES == 1) v += ((const float*)Rp)[(size_t)gm * N + gn];
        if (RES == 2) v += bf2f(((const __bf16*)Rp)[(size_t)gm * N + gn]);
        C[(size_t)gm * N + gn] = f2bf(v);
      }
    }
  }
}

// ---------------- pooled keys + LN0 ----------------
__global__ __launch_bounds__(256)
void pool_ln0_kernel(const float* __restrict__ x, const float* __restrict__ Wp,
                     const float* __restrict__ bp, const float* __restrict__ g0,
                     const float* __restrict__ be0, __bf16* __restrict__ xp) {
  const int rowid = blockIdx.x;          // bv*16 + t
  const int t  = rowid & 15;
  const int bv = rowid >> 4;
  const int tid = threadIdx.x;

  float w[16];
#pragma unroll
  for (int g = 0; g < 16; ++g) w[g] = Wp[g];
  const float bpv = bp[0];

  const float* base = x + ((size_t)bv * PQ + t) * DQ;
  float a0 = bpv, a1 = bpv;
#pragma unroll
  for (int g = 0; g < 16; ++g) {
    const float* r = base + (size_t)g * 16 * DQ;
    a0 += r[tid] * w[g];
    a1 += r[tid + 256] * w[g];
  }
  float s = a0 + a1, ss = a0 * a0 + a1 * a1;
#pragma unroll
  for (int off = 16; off > 0; off >>= 1) {
    s  += __shfl_xor(s, off, 32);
    ss += __shfl_xor(ss, off, 32);
  }
  __shared__ float rs[8], rss[8];
  if ((tid & 31) == 0) { rs[tid >> 5] = s; rss[tid >> 5] = ss; }
  __syncthreads();
  float S = 0.f, SS = 0.f;
#pragma unroll
  for (int wv = 0; wv < 8; ++wv) { S += rs[wv]; SS += rss[wv]; }
  const float mean = S * (1.f / DQ);
  const float inv  = rsqrtf(SS * (1.f / DQ) - mean * mean + 1e-5f);
  __bf16* out = xp + (size_t)rowid * DQ;
  out[tid]       = f2bf((a0 - mean) * inv * g0[tid] + be0[tid]);
  out[tid + 256] = f2bf((a1 - mean) * inv * g0[tid + 256] + be0[tid + 256]);
}

// ---------------- generic LayerNorm over D=512 ----------------
template<int OUT_F32>
__global__ __launch_bounds__(256)
void ln_kernel(const __bf16* __restrict__ X, const float* __restrict__ g,
               const float* __restrict__ be, void* __restrict__ Y) {
  const int row = blockIdx.x;
  const int tid = threadIdx.x;
  const __bf16* x = X + (size_t)row * DQ;
  const float v0 = bf2f(x[tid]);
  const float v1 = bf2f(x[tid + 256]);
  float s = v0 + v1, ss = v0 * v0 + v1 * v1;
#pragma unroll
  for (int off = 16; off > 0; off >>= 1) {
    s  += __shfl_xor(s, off, 32);
    ss += __shfl_xor(ss, off, 32);
  }
  __shared__ float rs[8], rss[8];
  if ((tid & 31) == 0) { rs[tid >> 5] = s; rss[tid >> 5] = ss; }
  __syncthreads();
  float S = 0.f, SS = 0.f;
#pragma unroll
  for (int wv = 0; wv < 8; ++wv) { S += rs[wv]; SS += rss[wv]; }
  const float mean = S * (1.f / DQ);
  const float inv  = rsqrtf(SS * (1.f / DQ) - mean * mean + 1e-5f);
  const float o0 = (v0 - mean) * inv * g[tid] + be[tid];
  const float o1 = (v1 - mean) * inv * g[tid + 256] + be[tid + 256];
  if (OUT_F32) {
    float* y = (float*)Y + (size_t)row * DQ;
    y[tid] = o0; y[tid + 256] = o1;
  } else {
    __bf16* y = (__bf16*)Y + (size_t)row * DQ;
    y[tid] = f2bf(o0); y[tid + 256] = f2bf(o1);
  }
}

// ---------------- patch-level attention: O = softmax(scale*Q K^T) K ----------------
__global__ __launch_bounds__(256)
void attn_patch_kernel(const __bf16* __restrict__ Q, const __bf16* __restrict__ Kv,
                       __bf16* __restrict__ O) {
  __shared__ __bf16 Ks[16][DQ];
  const int bv  = blockIdx.y;
  const int tid = threadIdx.x;
  {
    const v8bf* src = (const v8bf*)(Kv + (size_t)bv * 16 * DQ);
    v8bf* dst = (v8bf*)&Ks[0][0];
#pragma unroll
    for (int i = 0; i < 4; ++i) dst[tid + 256 * i] = src[tid + 256 * i];
  }
  __syncthreads();

  const int wave = tid >> 5, lane = tid & 31;
  const int p = blockIdx.x * 8 + wave;
  const size_t row = (size_t)bv * PQ + p;

  float ql[16];
  {
    const v8bf* qp = (const v8bf*)(Q + row * DQ + lane * 16);
    v8bf q0 = qp[0], q1 = qp[1];
#pragma unroll
    for (int i = 0; i < 8; ++i) { ql[i] = bf2f(q0[i]); ql[8 + i] = bf2f(q1[i]); }
  }

  float s[16];
#pragma unroll
  for (int j = 0; j < 16; ++j) {
    const v8bf* kp = (const v8bf*)&Ks[j][lane * 16];
    v8bf k0 = kp[0], k1 = kp[1];
    float ps = 0.f;
#pragma unroll
    for (int i = 0; i < 8; ++i) ps += ql[i] * bf2f(k0[i]) + ql[8 + i] * bf2f(k1[i]);
#pragma unroll
    for (int off = 16; off > 0; off >>= 1) ps += __shfl_xor(ps, off, 32);
    s[j] = ps * ATT_SCALE;
  }
  float mx = s[0];
#pragma unroll
  for (int j = 1; j < 16; ++j) mx = fmaxf(mx, s[j]);
  float den = 0.f, a[16];
#pragma unroll
  for (int j = 0; j < 16; ++j) { a[j] = expf(s[j] - mx); den += a[j]; }
  const float dinv = 1.f / den;
  float o[16] = {};
#pragma unroll
  for (int j = 0; j < 16; ++j) {
    const float aj = a[j] * dinv;
    const v8bf* kp = (const v8bf*)&Ks[j][lane * 16];
    v8bf k0 = kp[0], k1 = kp[1];
#pragma unroll
    for (int i = 0; i < 8; ++i) { o[i] += aj * bf2f(k0[i]); o[8 + i] += aj * bf2f(k1[i]); }
  }
  __bf16 ob[16];
#pragma unroll
  for (int i = 0; i < 16; ++i) ob[i] = f2bf(o[i]);
  v8bf* op = (v8bf*)(O + row * DQ + lane * 16);
  op[0] = *(v8bf*)&ob[0];
  op[1] = *(v8bf*)&ob[8];
}

// ---------------- variable-level attention + concat ----------------
__global__ __launch_bounds__(256)
void attn_var_kernel(const __bf16* __restrict__ Q2, const __bf16* __restrict__ K2,
                     const int* __restrict__ ccc, __bf16* __restrict__ Cc) {
  const int tid = threadIdx.x, wave = tid >> 5, lane = tid & 31;
  const size_t tok = (size_t)blockIdx.x * 8 + wave;   // (b*V + v)*P + p
  const int p = (int)(tok & 255);
  const int v = (int)((tok >> 8) & 31);
  const int b = (int)(tok >> 13);

  v8bf* outp = (v8bf*)(Cc + tok * DQ + lane * 16);
  if (p < 128) {
    const v8bf* kp = (const v8bf*)(K2 + tok * DQ + lane * 16);
    outp[0] = kp[0]; outp[1] = kp[1];
    return;
  }
  float ql[16];
  {
    const v8bf* qp = (const v8bf*)(Q2 + tok * DQ + lane * 16);
    v8bf q0 = qp[0], q1 = qp[1];
#pragma unroll
    for (int i = 0; i < 8; ++i) { ql[i] = bf2f(q0[i]); ql[8 + i] = bf2f(q1[i]); }
  }
  size_t krow[8];
#pragma unroll
  for (int n = 0; n < 8; ++n) {
    const int vj = ccc[((b * VQ) + v) * 8 + n];
    krow[n] = ((size_t)(b * VQ + vj) * PQ + p) * DQ;
  }
  float s[8];
#pragma unroll
  for (int n = 0; n < 8; ++n) {
    const v8bf* kp = (const v8bf*)(K2 + krow[n] + lane * 16);
    v8bf k0 = kp[0], k1 = kp[1];
    float ps = 0.f;
#pragma unroll
    for (int i = 0; i < 8; ++i) ps += ql[i] * bf2f(k0[i]) + ql[8 + i] * bf2f(k1[i]);
#pragma unroll
    for (int off = 16; off > 0; off >>= 1) ps += __shfl_xor(ps, off, 32);
    s[n] = ps * ATT_SCALE;
  }
  float mx = s[0];
#pragma unroll
  for (int n = 1; n < 8; ++n) mx = fmaxf(mx, s[n]);
  float den = 0.f, a[8];
#pragma unroll
  for (int n = 0; n < 8; ++n) { a[n] = expf(s[n] - mx); den += a[n]; }
  const float dinv = 1.f / den;
  float o[16] = {};
#pragma unroll
  for (int n = 0; n < 8; ++n) {
    const float an = a[n] * dinv;
    const v8bf* kp = (const v8bf*)(K2 + krow[n] + lane * 16);
    v8bf k0 = kp[0], k1 = kp[1];
#pragma unroll
    for (int i = 0; i < 8; ++i) { o[i] += an * bf2f(k0[i]); o[8 + i] += an * bf2f(k1[i]); }
  }
  __bf16 ob[16];
#pragma unroll
  for (int i = 0; i < 16; ++i) ob[i] = f2bf(o[i]);
  outp[0] = *(v8bf*)&ob[0];
  outp[1] = *(v8bf*)&ob[8];
}

// ---------------- host-side launch ----------------
extern "C" void kernel_launch(void* const* d_in, const int* in_sizes, int n_in,
                              void* d_out, int out_size, void* d_ws, size_t ws_size,
                              hipStream_t stream) {
  (void)in_sizes; (void)n_in; (void)out_size; (void)ws_size;
  const float* x    = (const float*)d_in[0];
  const int*   vccc = (const int*)  d_in[1];
  const float* Wq1 = (const float*)d_in[2],  *bq1 = (const float*)d_in[3];
  const float* Wk1 = (const float*)d_in[4],  *bk1 = (const float*)d_in[5];
  const float* Wo1 = (const float*)d_in[6],  *bo1 = (const float*)d_in[7];
  const float* Wp  = (const float*)d_in[8],  *bp  = (const float*)d_in[9];
  const float* Wq2 = (const float*)d_in[10], *bq2 = (const float*)d_in[11];
  const float* Wk2 = (const float*)d_in[12], *bk2 = (const float*)d_in[13];
  const float* Wo2 = (const float*)d_in[14], *bo2 = (const float*)d_in[15];
  const float* g0 = (const float*)d_in[16], *be0 = (const float*)d_in[17];
  const float* g1 = (const float*)d_in[18], *be1 = (const float*)d_in[19];
  const float* g2 = (const float*)d_in[20], *be2 = (const float*)d_in[21];
  const float* g3 = (const float*)d_in[22], *be3 = (const float*)d_in[23];
  const float* g4 = (const float*)d_in[24], *be4 = (const float*)d_in[25];
  const float* W1f = (const float*)d_in[26], *b1f = (const float*)d_in[27];
  const float* W2f = (const float*)d_in[28], *b2f = (const float*)d_in[29];
  const float* W3f = (const float*)d_in[30], *b3f = (const float*)d_in[31];
  const float* W4f = (const float*)d_in[32], *b4f = (const float*)d_in[33];

  char* ws = (char*)d_ws;
  size_t off = 0;
  auto take = [&](size_t bytes) -> char* {
    char* p = ws + off;
    off += (bytes + 255) & ~(size_t)255;
    return p;
  };
  __bf16* wq1b = (__bf16*)take((size_t)DQ * DQ * 2);
  __bf16* wk1b = (__bf16*)take((size_t)DQ * DQ * 2);
  __bf16* wo1b = (__bf16*)take((size_t)DQ * DQ * 2);
  __bf16* wq2b = (__bf16*)take((size_t)DQ * DQ * 2);
  __bf16* wk2b = (__bf16*)take((size_t)DQ * DQ * 2);
  __bf16* wo2b = (__bf16*)take((size_t)DQ * DQ * 2);
  __bf16* w1fb = (__bf16*)take((size_t)2 * DQ * DQ * 2);
  __bf16* w2fb = (__bf16*)take((size_t)DQ * 2 * DQ * 2);
  __bf16* w3fb = (__bf16*)take((size_t)2 * DQ * DQ * 2);
  __bf16* w4fb = (__bf16*)take((size_t)DQ * 2 * DQ * 2);
  __bf16* xp   = (__bf16*)take((size_t)MPQ * DQ * 2);
  __bf16* k1   = (__bf16*)take((size_t)MPQ * DQ * 2);
  __bf16* P0   = (__bf16*)take((size_t)MQ * DQ * 2);
  __bf16* P1   = (__bf16*)take((size_t)MQ * DQ * 2);
  __bf16* P2   = (__bf16*)take((size_t)MQ * DQ * 2);
  __bf16* Hb   = (__bf16*)take((size_t)MQ * 2 * DQ * 2);  // FFN hidden / concat buffer

  const int nSq = DQ * DQ, nFf = 2 * DQ * DQ;
  conv_f32_bf16_kernel<<<nSq / 1024, 256, 0, stream>>>(Wq1, wq1b);
  conv_f32_bf16_kernel<<<nSq / 1024, 256, 0, stream>>>(Wk1, wk1b);
  conv_f32_bf16_kernel<<<nSq / 1024, 256, 0, stream>>>(Wo1, wo1b);
  conv_f32_bf16_kernel<<<nSq / 1024, 256, 0, stream>>>(Wq2, wq2b);
  conv_f32_bf16_kernel<<<nSq / 1024, 256, 0, stream>>>(Wk2, wk2b);
  conv_f32_bf16_kernel<<<nSq / 1024, 256, 0, stream>>>(Wo2, wo2b);
  conv_f32_bf16_kernel<<<nFf / 1024, 256, 0, stream>>>(W1f, w1fb);
  conv_f32_bf16_kernel<<<nFf / 1024, 256, 0, stream>>>(W2f, w2fb);
  conv_f32_bf16_kernel<<<nFf / 1024, 256, 0, stream>>>(W3f, w3fb);
  conv_f32_bf16_kernel<<<nFf / 1024, 256, 0, stream>>>(W4f, w4fb);

  const dim3 gBig(MQ / 128, DQ / 128);        // 512 x 4
  const dim3 gBigW(MQ / 128, (2 * DQ) / 128); // 512 x 8
  const dim3 gSm(MPQ / 128, DQ / 128);        // 32 x 4

  // 1. pooled keys + LN0
  pool_ln0_kernel<<<MPQ, 256, 0, stream>>>(x, Wp, bp, g0, be0, xp);
  // 2. k = xp @ Wk1^T + bk1
  gemm_bf16_kernel<0, 0, 0><<<gSm, 256, 0, stream>>>(xp, wk1b, bk1, nullptr, k1, MPQ, DQ, DQ);
  // 3. q = x @ Wq1^T + bq1   (f32 A converted in-flight)
  gemm_bf16_kernel<1, 0, 0><<<gBig, 256, 0, stream>>>(x, wq1b, bq1, nullptr, P0, MQ, DQ, DQ);
  // 4. patch attention -> P1
  attn_patch_kernel<<<dim3(PQ / 8, BQ * VQ), 256, 0, stream>>>(P0, k1, P1);
  // 5. a1 @ Wo1 + bo1 + x  -> P2
  gemm_bf16_kernel<0, 0, 1><<<gBig, 256, 0, stream>>>(P1, wo1b, bo1, x, P2, MQ, DQ, DQ);
  // 6. x1 = LN1 -> P0
  ln_kernel<0><<<MQ, 256, 0, stream>>>(P2, g1, be1, P0);
  // 7. H = gelu(x1 @ W1f + b1f) -> Hb
  gemm_bf16_kernel<0, 1, 0><<<gBigW, 256, 0, stream>>>(P0, w1fb, b1f, nullptr, Hb, MQ, 2 * DQ, DQ);
  // 8. x2 = x1 + H @ W2f + b2f -> P1
  gemm_bf16_kernel<0, 0, 2><<<gBig, 256, 0, stream>>>(Hb, w2fb, b2f, P0, P1, MQ, DQ, 2 * DQ);
  // 9. xc = LN2 -> P2   (layout stays [B,V,P,D]; transpose is pure indexing)
  ln_kernel<0><<<MQ, 256, 0, stream>>>(P1, g2, be2, P2);
  // 10. K2 = xc @ Wk2 + bk2 -> P0
  gemm_bf16_kernel<0, 0, 0><<<gBig, 256, 0, stream>>>(P2, wk2b, bk2, nullptr, P0, MQ, DQ, DQ);
  // 11. Q2 = xc @ Wq2 + bq2 -> P1
  gemm_bf16_kernel<0, 0, 0><<<gBig, 256, 0, stream>>>(P2, wq2b, bq2, nullptr, P1, MQ, DQ, DQ);
  // 12. variable attention + concat -> Hb (as [M,512] bf16)
  attn_var_kernel<<<MQ / 8, 256, 0, stream>>>(P1, P0, vccc, Hb);
  // 13. a2o = concat @ Wo2 + bo2 + xc -> P0
  gemm_bf16_kernel<0, 0, 2><<<gBig, 256, 0, stream>>>(Hb, wo2b, bo2, P2, P0, MQ, DQ, DQ);
  // 14. x3 = LN3 -> P1
  ln_kernel<0><<<MQ, 256, 0, stream>>>(P0, g3, be3, P1);
  // 15. H2 = gelu(x3 @ W3f + b3f) -> Hb
  gemm_bf16_kernel<0, 1, 0><<<gBigW, 256, 0, stream>>>(P1, w3fb, b3f, nullptr, Hb, MQ, 2 * DQ, DQ);
  // 16. x4 = x3 + H2 @ W4f + b4f -> P2
  gemm_bf16_kernel<0, 0, 2><<<gBig, 256, 0, stream>>>(Hb, w4fb, b4f, P1, P2, MQ, DQ, 2 * DQ);
  // 17. out = LN4 -> d_out (f32, already [B,V,P,D])
  ln_kernel<1><<<MQ, 256, 0, stream>>>(P2, g4, be4, d_out);
}
